// KernLM_25606595019377
// MI455X (gfx1250) — matrix operand order repaired and checked by reading, not compiled, not run on hardware
//
#include <hip/hip_runtime.h>

// ---------------- model constants (match reference) ----------------
#define TT   256      // sequence length
#define DD   256      // embedding dim
#define HH   4        // heads
#define HDIM 64       // head dim
#define LL   8        // layers
#define VV   512      // vocab
#define BB   128      // batch
#define MM   (BB*TT)  // 32768 token rows
#define KK   256      // contraction dim of every GEMM in this model

typedef unsigned short u16;
typedef __attribute__((ext_vector_type(16))) __bf16        v16bf;
typedef __attribute__((ext_vector_type(8)))  float         v8f;
typedef __attribute__((ext_vector_type(4)))  unsigned int  u32x4;

union Frag {
  v16bf bf;
  u16   u[16];
  u32x4 q[2];
};

__device__ __forceinline__ u16 bf16r(float f) {
  unsigned int u = __float_as_uint(f);
  u += 0x7FFFu + ((u >> 16) & 1u);           // round-to-nearest-even
  return (u16)(u >> 16);
}

__device__ __forceinline__ v8f vzero8() {
  v8f z;
#pragma unroll
  for (int i = 0; i < 8; i++) z[i] = 0.0f;
  return z;
}

__device__ __forceinline__ v8f wmma_bf16(v16bf a, v16bf b, v8f c) {
  // D = A(16x32 bf16) * B(32x16 bf16) + C(16x16 f32)
  return __builtin_amdgcn_wmma_f32_16x16x32_bf16(false, a, false, b,
                                                 (short)0, c, false, false);
}

// ---------------- embedding + sinusoidal positional encoding ----------------
__global__ __launch_bounds__(DD) void k_embed(const int* __restrict__ idx,
                                              const float* __restrict__ emb,
                                              float* __restrict__ x) {
  int m = blockIdx.x;            // token row 0..MM-1
  int d = threadIdx.x;           // 0..255
  int t = m & (TT - 1);
  int tok = idx[m];
  float freq = __expf(-(float)(d & ~1) * (9.210340371976184f / (float)DD));
  float ang  = (float)t * freq;
  float pe   = (d & 1) ? __cosf(ang) : __sinf(ang);
  x[(size_t)m * DD + d] = emb[(size_t)tok * DD + d] + pe;
}

// ---------------- f32 [K,N] -> bf16 transposed [N,K] ----------------
__global__ __launch_bounds__(256) void k_cvtT(const float* __restrict__ W,
                                              u16* __restrict__ WT,
                                              int K, int N) {
  int i = blockIdx.x * 256 + threadIdx.x;
  if (i >= K * N) return;
  int k = i / N, n = i - k * N;
  WT[(size_t)n * K + k] = bf16r(W[i]);
}

// ---------------- LayerNorm over D=256, one wave32 per token, bf16 out ----------------
__global__ __launch_bounds__(256) void k_ln(const float* __restrict__ x,
                                            const float* __restrict__ gamma,
                                            const float* __restrict__ beta,
                                            u16* __restrict__ out) {
  int lane = threadIdx.x & 31;
  int m = blockIdx.x * 8 + (threadIdx.x >> 5);
  const float* row = x + (size_t)m * DD;
  float v[8];
#pragma unroll
  for (int j = 0; j < 8; j++) v[j] = row[lane * 8 + j];
  float s = 0.f;
#pragma unroll
  for (int j = 0; j < 8; j++) s += v[j];
#pragma unroll
  for (int off = 16; off >= 1; off >>= 1) s += __shfl_xor(s, off, 32);
  float mu = s * (1.0f / (float)DD);
  float vs = 0.f;
#pragma unroll
  for (int j = 0; j < 8; j++) { float d = v[j] - mu; vs += d * d; }
#pragma unroll
  for (int off = 16; off >= 1; off >>= 1) vs += __shfl_xor(vs, off, 32);
  float inv = rsqrtf(vs * (1.0f / (float)DD) + 1e-5f);
#pragma unroll
  for (int j = 0; j < 8; j++) {
    int c = lane * 8 + j;
    float y = (v[j] - mu) * inv * gamma[c] + beta[c];
    out[(size_t)m * DD + c] = bf16r(y);
  }
}

// ---------------- WMMA GEMM: C[M,N] = A[M,K=256](bf16) * WT[N,K=256](bf16) ----------
// Block stages its 64x256 WT slab (32 KB) into LDS with CDNA5 async-to-LDS DMA.
// 8 waves; each wave computes a 32x64 C tile (two M-subtiles share B fragments:
// 8 WMMA per 4 LDS reads). K-loop fully unrolled (K == 256). Fused epilogue:
// +bias, relu, +f32 residual; f32 and/or bf16 out; optional transposed-V store.
__global__ __launch_bounds__(256) void k_gemm(const u16* __restrict__ A,
                                              const u16* __restrict__ WT,
                                              const float* __restrict__ bias,
                                              const float* __restrict__ resid,
                                              float* __restrict__ outF,
                                              u16* __restrict__ outB,
                                              u16* __restrict__ outVT,
                                              int M, int N, int relu) {
  __shared__ __align__(16) u16 ldsW[64][KK];    // 32 KB weight slab

  int lane = threadIdx.x & 31;
  int w    = threadIdx.x >> 5;
  int g    = lane >> 4;        // lane half
  int ln   = lane & 15;
  int m0   = blockIdx.x * 256 + w * 32;
  int n0   = blockIdx.y * 64;

  // ---- async stage of WT[n0..n0+63][0..255] (contiguous 32 KB) into LDS ----
  {
    const u16* slab = WT + (size_t)n0 * KK;
    unsigned ldsbase = (unsigned)(size_t)&ldsW[0][0];
    int tid = threadIdx.x;
#pragma unroll
    for (int it = 0; it < 8; it++) {
      int c = it * 256 + tid;                       // 16-byte chunk id, 0..2047
      unsigned loff = ldsbase + (unsigned)c * 16u;
      unsigned long long ga = (unsigned long long)(size_t)slab +
                              (unsigned long long)c * 16ull;
      asm volatile("global_load_async_to_lds_b128 %0, %1, off"
                   :: "v"(loff), "v"(ga) : "memory");
    }
    asm volatile("s_wait_asynccnt 0" ::: "memory");
    __syncthreads();
  }

  const u16* arow0 = A + (size_t)(m0 + ln) * KK;
  const u16* arow1 = A + (size_t)(m0 + 16 + ln) * KK;

  v8f acc[2][4];
#pragma unroll
  for (int mi = 0; mi < 2; mi++)
#pragma unroll
    for (int nt = 0; nt < 4; nt++) acc[mi][nt] = vzero8();

#pragma unroll
  for (int kb = 0; kb < KK; kb += 32) {
    Frag a0, a1;
    a0.q[0] = *(const u32x4*)(arow0 + kb + 16 * g);
    a0.q[1] = *(const u32x4*)(arow0 + kb + 16 * g + 8);
    a1.q[0] = *(const u32x4*)(arow1 + kb + 16 * g);
    a1.q[1] = *(const u32x4*)(arow1 + kb + 16 * g + 8);
#pragma unroll
    for (int nt = 0; nt < 4; nt++) {
      const u16* brow = &ldsW[nt * 16 + ln][kb + 16 * g];
      Frag b;
      b.q[0] = *(const u32x4*)(brow);
      b.q[1] = *(const u32x4*)(brow + 8);
      acc[0][nt] = wmma_bf16(a0.bf, b.bf, acc[0][nt]);
      acc[1][nt] = wmma_bf16(a1.bf, b.bf, acc[1][nt]);
    }
  }

#pragma unroll
  for (int mi = 0; mi < 2; mi++) {
#pragma unroll
    for (int nt = 0; nt < 4; nt++) {
      int n = n0 + nt * 16 + ln;
      float bv = bias ? bias[n] : 0.0f;
#pragma unroll
      for (int r = 0; r < 8; r++) {
        int m = m0 + mi * 16 + 8 * g + r;
        float val = acc[mi][nt][r] + bv;
        if (relu) val = fmaxf(val, 0.0f);
        if (resid) val += resid[(size_t)m * N + n];
        if (outF) outF[(size_t)m * N + n] = val;
        if (outB) outB[(size_t)m * N + n] = bf16r(val);
        if (outVT) {                                // [B][H][HDIM][TT]
          int bb = m >> 8, t = m & (TT - 1);
          int hh = n >> 6, hd = n & (HDIM - 1);
          outVT[(((size_t)bb * HH + hh) * HDIM + hd) * TT + t] = bf16r(val);
        }
      }
    }
  }
}

// ---------------- fused causal attention, one (b,h) per blockIdx.x ----------------
// block = 4 waves (128 thr); wave owns a 16-row q strip; grid.y covers T/64.
// VT is bf16 in [B][H][HDIM][TT] layout, so P@V B-fragments load contiguously.
__global__ __launch_bounds__(128) void k_attn(const u16* __restrict__ Q,
                                              const u16* __restrict__ Kc,
                                              const u16* __restrict__ VT,
                                              u16* __restrict__ O) {
  __shared__ __align__(16) u16 ldsP[4][16][TT];   // 32 KB: bf16 P strip per wave

  int lane = threadIdx.x & 31;
  int w    = threadIdx.x >> 5;
  int g    = lane >> 4;
  int ln   = lane & 15;
  int b    = blockIdx.x >> 2;          // HH == 4
  int h    = blockIdx.x & 3;
  int qbase = blockIdx.y * 64 + w * 16;
  size_t rowb = (size_t)b * TT;

  // Q strip A-fragments (HD = 64 -> two K=32 fragments)
  Frag aq[2];
  const u16* qrow = Q + (rowb + qbase + ln) * DD + h * HDIM;
#pragma unroll
  for (int c = 0; c < 2; c++) {
    aq[c].q[0] = *(const u32x4*)(qrow + c * 32 + 16 * g);
    aq[c].q[1] = *(const u32x4*)(qrow + c * 32 + 16 * g + 8);
  }

  // scores: 16 key tiles of 16
  v8f s[16];
#pragma unroll
  for (int nt = 0; nt < 16; nt++) {
    const u16* krow = Kc + (rowb + nt * 16 + ln) * DD + h * HDIM + 16 * g;
    v8f acc = vzero8();
#pragma unroll
    for (int c = 0; c < 2; c++) {
      Frag bq;
      bq.q[0] = *(const u32x4*)(krow + c * 32);
      bq.q[1] = *(const u32x4*)(krow + c * 32 + 8);
      acc = wmma_bf16(aq[c].bf, bq.bf, acc);
    }
    s[nt] = acc;
  }

  // causal mask + scale (reference scales by D^-0.5 = 1/16)
  const float scale = 0.0625f;
#pragma unroll
  for (int nt = 0; nt < 16; nt++) {
#pragma unroll
    for (int r = 0; r < 8; r++) {
      int qr = qbase + 8 * g + r;
      int kc = nt * 16 + ln;
      float val = s[nt][r] * scale;
      if (kc > qr) val = -1e30f;
      s[nt][r] = val;
    }
  }

  // softmax across 256 keys: per-lane partial + shuffle within 16-lane half
  float mx[8], inv[8];
#pragma unroll
  for (int r = 0; r < 8; r++) {
    float m_ = -1e30f;
#pragma unroll
    for (int nt = 0; nt < 16; nt++) m_ = fmaxf(m_, s[nt][r]);
#pragma unroll
    for (int off = 8; off >= 1; off >>= 1) m_ = fmaxf(m_, __shfl_xor(m_, off, 32));
    mx[r] = m_;
  }
#pragma unroll
  for (int r = 0; r < 8; r++) {
    float t = 0.f;
#pragma unroll
    for (int nt = 0; nt < 16; nt++) {
      float p = __expf(s[nt][r] - mx[r]);
      s[nt][r] = p;
      t += p;
    }
#pragma unroll
    for (int off = 8; off >= 1; off >>= 1) t += __shfl_xor(t, off, 32);
    inv[r] = 1.0f / t;
  }

  // spill P (bf16) through LDS: C-layout -> A-layout
#pragma unroll
  for (int nt = 0; nt < 16; nt++)
#pragma unroll
    for (int r = 0; r < 8; r++)
      ldsP[w][8 * g + r][nt * 16 + ln] = bf16r(s[nt][r] * inv[r]);
  __syncthreads();

  // out[16,64] = P[16,256] @ V[256,64];  B-frag: VT row = head-dim column
  const u16* prow = &ldsP[w][ln][0];
#pragma unroll
  for (int ht = 0; ht < 4; ht++) {
    const u16* vtrow = VT + (((size_t)b * HH + h) * HDIM + (ht * 16 + ln)) * TT + 16 * g;
    v8f acc = vzero8();
#pragma unroll
    for (int kb = 0; kb < TT; kb += 32) {
      Frag pa;
      pa.q[0] = *(const u32x4*)(prow + kb + 16 * g);
      pa.q[1] = *(const u32x4*)(prow + kb + 16 * g + 8);
      Frag bv;
      bv.q[0] = *(const u32x4*)(vtrow + kb);
      bv.q[1] = *(const u32x4*)(vtrow + kb + 8);
      acc = wmma_bf16(pa.bf, bv.bf, acc);
    }
#pragma unroll
    for (int r = 0; r < 8; r++)
      O[(rowb + qbase + 8 * g + r) * DD + h * HDIM + ht * 16 + ln] = bf16r(acc[r]);
  }
}

// ---------------- host orchestration ----------------
extern "C" void kernel_launch(void* const* d_in, const int* in_sizes, int n_in,
                              void* d_out, int out_size, void* d_ws, size_t ws_size,
                              hipStream_t stream) {
  (void)in_sizes; (void)n_in; (void)out_size; (void)ws_size;

  const int*   idx     = (const int*)  d_in[0];
  const float* tok_emb = (const float*)d_in[1];
  const float* Wq      = (const float*)d_in[2];
  const float* Wk      = (const float*)d_in[3];
  const float* Wv      = (const float*)d_in[4];
  const float* Wo      = (const float*)d_in[5];
  const float* bo      = (const float*)d_in[6];
  const float* W1      = (const float*)d_in[7];
  const float* b1      = (const float*)d_in[8];
  const float* W2      = (const float*)d_in[9];
  const float* b2      = (const float*)d_in[10];
  const float* g1      = (const float*)d_in[11];
  const float* be1     = (const float*)d_in[12];
  const float* g2      = (const float*)d_in[13];
  const float* be2     = (const float*)d_in[14];
  const float* gf      = (const float*)d_in[15];
  const float* bff     = (const float*)d_in[16];
  const float* Wh      = (const float*)d_in[17];
  const float* bh      = (const float*)d_in[18];

  // workspace layout: [bf16 WT weights | x f32 (M*D) | h bf16 (M*D)]
  u16*   wt   = (u16*)d_ws;
  float* x    = (float*)((char*)d_ws + 8u * 1024u * 1024u);
  u16*   hbuf = (u16*)  ((char*)d_ws + 8u * 1024u * 1024u + (size_t)MM * DD * 4);

  // d_out doubles as q/k/v + LN2/ReLU scratch (head GEMM overwrites all of it)
  u16* qbuf = (u16*)d_out;
  u16* kbuf = qbuf + (size_t)MM * DD;
  u16* vbuf = kbuf + (size_t)MM * DD;   // holds V in transposed [B][H][HD][T] layout

  const size_t DSQ = (size_t)DD * DD;   // 65536
  u16* wth = wt + (size_t)LL * 6 * DSQ;

  // 1) convert + transpose all weights to bf16 [N,K]
  {
    dim3 grid((DD * DD + 255) / 256);
    for (int l = 0; l < LL; l++) {
      k_cvtT<<<grid, 256, 0, stream>>>(Wq + l * DSQ, wt + ((size_t)l * 6 + 0) * DSQ, DD, DD);
      k_cvtT<<<grid, 256, 0, stream>>>(Wk + l * DSQ, wt + ((size_t)l * 6 + 1) * DSQ, DD, DD);
      k_cvtT<<<grid, 256, 0, stream>>>(Wv + l * DSQ, wt + ((size_t)l * 6 + 2) * DSQ, DD, DD);
      k_cvtT<<<grid, 256, 0, stream>>>(Wo + l * DSQ, wt + ((size_t)l * 6 + 3) * DSQ, DD, DD);
      k_cvtT<<<grid, 256, 0, stream>>>(W1 + l * DSQ, wt + ((size_t)l * 6 + 4) * DSQ, DD, DD);
      k_cvtT<<<grid, 256, 0, stream>>>(W2 + l * DSQ, wt + ((size_t)l * 6 + 5) * DSQ, DD, DD);
    }
    k_cvtT<<<dim3((DD * VV + 255) / 256), 256, 0, stream>>>(Wh, wth, DD, VV);
  }

  // 2) embedding + positional encoding
  k_embed<<<MM, DD, 0, stream>>>(idx, tok_emb, x);

  dim3 ggemm(MM / 256, DD / 64);
  dim3 ghead(MM / 256, VV / 64);
  dim3 gattn(BB * HH, TT / 64);
  dim3 gln(MM / 8);

  // 3) transformer layers
  for (int l = 0; l < LL; l++) {
    const u16* wq = wt + ((size_t)l * 6 + 0) * DSQ;
    const u16* wk = wt + ((size_t)l * 6 + 1) * DSQ;
    const u16* wv = wt + ((size_t)l * 6 + 2) * DSQ;
    const u16* wo = wt + ((size_t)l * 6 + 3) * DSQ;
    const u16* w1 = wt + ((size_t)l * 6 + 4) * DSQ;
    const u16* w2 = wt + ((size_t)l * 6 + 5) * DSQ;

    k_ln<<<gln, 256, 0, stream>>>(x, g1 + l * DD, be1 + l * DD, hbuf);

    k_gemm<<<ggemm, 256, 0, stream>>>(hbuf, wq, nullptr, nullptr, nullptr, qbuf, nullptr, MM, DD, 0);
    k_gemm<<<ggemm, 256, 0, stream>>>(hbuf, wk, nullptr, nullptr, nullptr, kbuf, nullptr, MM, DD, 0);
    k_gemm<<<ggemm, 256, 0, stream>>>(hbuf, wv, nullptr, nullptr, nullptr, nullptr, vbuf, MM, DD, 0);

    k_attn<<<gattn, 128, 0, stream>>>(qbuf, kbuf, vbuf, hbuf);

    // x = x + att @ Wo + bo
    k_gemm<<<ggemm, 256, 0, stream>>>(hbuf, wo, bo + l * DD, x, x, nullptr, nullptr, MM, DD, 0);

    k_ln<<<gln, 256, 0, stream>>>(x, g2 + l * DD, be2 + l * DD, qbuf);
    // relu(h @ W1 + b1)
    k_gemm<<<ggemm, 256, 0, stream>>>(qbuf, w1, b1 + l * DD, nullptr, nullptr, kbuf, nullptr, MM, DD, 1);
    // x = x + (...) @ W2 + b2
    k_gemm<<<ggemm, 256, 0, stream>>>(kbuf, w2, b2 + l * DD, x, x, nullptr, nullptr, MM, DD, 0);
  }

  // 4) final LN + head
  k_ln<<<gln, 256, 0, stream>>>(x, gf, bff, hbuf);
  k_gemm<<<ghead, 256, 0, stream>>>(hbuf, wth, bh, nullptr, (float*)d_out, nullptr, nullptr, MM, VV, 0);
}
